// DehazingMVSNet_17282948399141
// MI455X (gfx1250) — matrix operand order repaired and checked by reading, not compiled, not run on hardware
//
#include <hip/hip_runtime.h>
#include <hip/hip_bf16.h>
#include <math.h>

typedef __attribute__((ext_vector_type(16))) _Float16 v16h;
typedef __attribute__((ext_vector_type(8)))  float    v8f;
typedef int v4i_gcc __attribute__((vector_size(16)));   // matches builtin proto

#define HH 192
#define WW 256
#define NSTEPS 256

#define NP0 49152   /* 192x256 */
#define NP1 12288   /* 96x128  */
#define NP2 3072    /* 48x64   */
#define NP3 768     /* 24x32   */
#define NP4 192     /* 12x16   */
#define NP5 48      /* 6x8     */

// CDNA5 async global->LDS staging (ASYNCcnt-tracked), with safe fallback.
#if defined(__gfx1250__) && \
    __has_builtin(__builtin_amdgcn_global_load_async_to_lds_b128) && \
    __has_builtin(__builtin_amdgcn_s_wait_asynccnt)
#define USE_ASYNC_LDS 1
#else
#define USE_ASYNC_LDS 0
#endif

static __device__ __forceinline__ unsigned short h2u(_Float16 v) {
  union { _Float16 f; unsigned short u; } c; c.f = v; return c.u;
}

// ---------------------------------------------------------------------------
// Cost volume: one thread per (depth, pixel). src/KRKiUV are small -> L2 hits.
// ---------------------------------------------------------------------------
__global__ __launch_bounds__(256) void cost_volume_kernel(
    const float* __restrict__ ref, const float* __restrict__ src,
    const float* __restrict__ KR,  const float* __restrict__ Kt,
    const float* __restrict__ Aamb, const float* __restrict__ beta,
    float* __restrict__ cv)
{
  const int HWp = HH * WW;
  long idx = (long)blockIdx.x * blockDim.x + threadIdx.x;
  if (idx >= (long)NSTEPS * HWp) return;
  int d  = (int)(idx / HWp);
  int hw = (int)(idx - (long)d * HWp);
  int h  = hw / WW;
  int w  = hw - h * WW;

  const float idmin = 1.0f / 50.0f, idmax = 1.0f / 0.5f;
  float id    = idmin + (float)d * ((idmax - idmin) / (float)(NSTEPS - 1));
  float depth = 1.0f / id;

  int pix = w * HH + h;                       // KRKiUV is u-major
  float Ktz = Kt[2];
  float px = depth * KR[0 * HWp + pix] + Kt[0];
  float py = depth * KR[1 * HWp + pix] + Kt[1];
  float pz = depth * KR[2 * HWp + pix] + Ktz;
  float inv = 1.0f / (pz + 1e-6f);
  float xs = px * inv - 0.5f;                 // grid->pixel collapses to u-0.5
  float ys = py * inv - 0.5f;
  float x0f = floorf(xs), y0f = floorf(ys);
  float wx = xs - x0f, wy = ys - y0f;

  float warped[3] = {0.f, 0.f, 0.f};
  float wds = 0.f;
#pragma unroll
  for (int jy = 0; jy < 2; ++jy) {
#pragma unroll
    for (int jx = 0; jx < 2; ++jx) {
      float xi = x0f + (float)jx;
      float yi = y0f + (float)jy;
      float valid = (xi >= 0.f && xi <= (float)(WW - 1) &&
                     yi >= 0.f && yi <= (float)(HH - 1)) ? 1.f : 0.f;
      int xc = (int)fminf(fmaxf(xi, 0.f), (float)(WW - 1));
      int yc = (int)fminf(fmaxf(yi, 0.f), (float)(HH - 1));
      float wt = ((jx == 0) ? (1.f - wx) : wx) * ((jy == 0) ? (1.f - wy) : wy);
      float vw = valid * wt;
      int off = yc * WW + xc;
      warped[0] += src[0 * HWp + off] * vw;
      warped[1] += src[1 * HWp + off] * vw;
      warped[2] += src[2 * HWp + off] * vw;
      float dv = depth * KR[2 * HWp + xc * HH + yc] + Ktz;  // demon image
      wds += dv * vw;
    }
  }
  float wd  = (wds == 0.0f) ? 0.0f : pz;
  float bet = beta[0];
  float eref = 1.0f / (expf(-depth * bet) + 1e-6f);
  float ewar = 1.0f / (expf(-wd * bet) + 1e-6f);
  float res = 0.f;
  bool bad = false;
#pragma unroll
  for (int c = 0; c < 3; ++c) {
    float Ac   = Aamb[c];
    float refj = (ref[c * HWp + hw] - Ac) * eref + Ac;
    float warj = (warped[c] - Ac) * ewar + Ac;
    res += fabsf(refj - warj);
    bad = bad || refj < 0.f || refj > 1.f || warj < 0.f || warj > 1.f;
  }
  cv[(size_t)d * HWp + hw] = bad ? 3.0f : res;
}

// ---------------------------------------------------------------------------
// f32 -> f16 conversion passes (run once per layer; bandwidth-trivial).
// Weights are packed with zero-padded rows (stride Kp = ceil32(K)) and
// zero-padded Cout (to a multiple of 128) so the GEMM A-stage is branch-free.
// ---------------------------------------------------------------------------
__global__ __launch_bounds__(256) void pack_weights_f16_kernel(
    const float* __restrict__ w, _Float16* __restrict__ w16,
    int Cout, int K, int Kp, long total)
{
  long i = (long)blockIdx.x * blockDim.x + threadIdx.x;
  if (i >= total) return;
  long row = i / Kp;
  int  col = (int)(i - row * Kp);
  w16[i] = (row < Cout && col < K) ? (_Float16)w[row * K + col] : (_Float16)0.f;
}

__global__ __launch_bounds__(256) void cvt_f16_kernel(
    const float* __restrict__ x, _Float16* __restrict__ y, long n)
{
  long i = (long)blockIdx.x * blockDim.x + threadIdx.x;
  if (i < n) y[i] = (_Float16)x[i];
}

// ---------------------------------------------------------------------------
// Implicit-GEMM convolution on the WMMA pipe (f16 in, f32 accumulate).
// M = Cout, N = Hout*Wout, K = Cin*KS*KS (Kp = padded).
// 256 threads = 8 wave32s; block tile 128(M) x 64(N) x 32(K); each wave owns a
// 32x32 patch -> 4x v_wmma_f32_16x16x32_f16 per k-step (2 A + 2 B fragments),
// amortizing the scattered im2col B-gather over 2x the FLOPs.
// Double-buffered LDS: one barrier per k-step, staging overlaps WMMA.
// Weight staging uses GLOBAL_LOAD_ASYNC_TO_LDS_B128 (ASYNCcnt).
// ---------------------------------------------------------------------------
#define CT_M 128
#define CT_N 64
#define CT_K 32
#define LDK  40   /* padded LDS row stride (halves): rows stay 16B aligned */

union V16U { v16h v; uint4 q[2]; };

template <int KS>
__global__ __launch_bounds__(256) void conv_igemm_wmma_kernel(
    const _Float16* __restrict__ x16, const _Float16* __restrict__ w16,
    float* __restrict__ y,
    int Cin, int Hin, int Win, int Hout, int Wout,
    int stride, int pad, int K, int Kp, int Cout)
{
  __shared__ _Float16 As[2][CT_M][LDK];
  __shared__ _Float16 Bs[2][CT_N][LDK];
  __shared__ int sOy[CT_N], sOx[CT_N];

  const int Np  = Hout * Wout;
  const int m0  = blockIdx.y * CT_M;
  const int n0  = blockIdx.x * CT_N;
  const int tid  = threadIdx.x;
  const int lane = tid & 31;
  const int wave = tid >> 5;
  const int wm = (wave & 3) * 32;    // M strip: 0,32,64,96
  const int wn = (wave >> 2) * 32;   // N strip: 0,32

  // Hoist output-pixel decode once per block.
  if (tid < CT_N) {
    int ng = n0 + tid;
    if (ng < Np) {
      int oy = ng / Wout;
      int ox = ng - oy * Wout;
      sOy[tid] = oy * stride - pad;
      sOx[tid] = ox * stride - pad;
    } else {
      sOy[tid] = -(1 << 28);
      sOx[tid] = -(1 << 28);
    }
  }
  __syncthreads();

  // Staging assignments.
  const int am  = tid >> 1;          // A row (0..127), 16 halves per thread
  const int ak0 = (tid & 1) * 16;    // A col chunk (two b128s)
  const int bn  = tid >> 2;          // B row (0..63), 8 halves per thread
  const int bk0 = (tid & 3) * 8;     // B col chunk
  const int oyb = sOy[bn];
  const int oxb = sOx[bn];
  const _Float16* wrow = w16 + (size_t)(m0 + am) * Kp + ak0;

  auto stageA = [&](int k0, int buf) {
#if USE_ASYNC_LDS
    __builtin_amdgcn_global_load_async_to_lds_b128(
        (v4i_gcc*)(void*)const_cast<_Float16*>(wrow + k0),
        (__attribute__((address_space(3))) v4i_gcc*)(void*)&As[buf][am][ak0],
        0, 0);
    __builtin_amdgcn_global_load_async_to_lds_b128(
        (v4i_gcc*)(void*)const_cast<_Float16*>(wrow + k0 + 8),
        (__attribute__((address_space(3))) v4i_gcc*)(void*)&As[buf][am][ak0 + 8],
        0, 0);
#else
    __builtin_prefetch(wrow + k0 + CT_K, 0, 1);
    *(uint4*)&As[buf][am][ak0]     = *(const uint4*)(wrow + k0);
    *(uint4*)&As[buf][am][ak0 + 8] = *(const uint4*)(wrow + k0 + 8);
#endif
  };

  auto stageB = [&](int k0, int buf) {
    unsigned short hv[8];
#pragma unroll
    for (int j = 0; j < 8; ++j) {
      int kg = k0 + bk0 + j;
      _Float16 v = (_Float16)0.f;
      if (kg < K) {
        int ci  = kg / (KS * KS);            // constant divides (KS templated)
        int rem = kg - ci * (KS * KS);
        int fy  = rem / KS;
        int fx  = rem - fy * KS;
        int iy  = oyb + fy;
        int ix  = oxb + fx;
        if ((unsigned)iy < (unsigned)Hin && (unsigned)ix < (unsigned)Win)
          v = x16[((size_t)ci * Hin + iy) * Win + ix];
      }
      hv[j] = h2u(v);
    }
    uint4 pk;
    pk.x = (unsigned)hv[0] | ((unsigned)hv[1] << 16);
    pk.y = (unsigned)hv[2] | ((unsigned)hv[3] << 16);
    pk.z = (unsigned)hv[4] | ((unsigned)hv[5] << 16);
    pk.w = (unsigned)hv[6] | ((unsigned)hv[7] << 16);
    *(uint4*)&Bs[buf][bn][bk0] = pk;
  };

  v8f acc00 = {0.f, 0.f, 0.f, 0.f, 0.f, 0.f, 0.f, 0.f};
  v8f acc01 = acc00, acc10 = acc00, acc11 = acc00;

  // Prologue: fill buffer 0.
  stageA(0, 0);
  stageB(0, 0);
#if USE_ASYNC_LDS
  __builtin_amdgcn_s_wait_asynccnt(0);
#endif
  __syncthreads();

  int cur = 0;
  for (int k0 = 0; k0 < Kp; k0 += CT_K) {
    // Overlap: stage the next tile into the other buffer while computing.
    int nk = k0 + CT_K;
    if (nk < Kp) {
      stageA(nk, cur ^ 1);
      stageB(nk, cur ^ 1);
    }

    // ---- fragments (wave32 ISA layouts), all 16B-aligned LDS loads ----
    const int r  = lane & 15;
    const int hi = lane >> 4;
    V16U a0, a1, b0, b1;
    a0.q[0] = *(const uint4*)&As[cur][wm + r][8 * hi];       // A: K = 8*hi+t
    a0.q[1] = *(const uint4*)&As[cur][wm + r][16 + 8 * hi];  //    K = 16+8*hi+t
    a1.q[0] = *(const uint4*)&As[cur][wm + 16 + r][8 * hi];
    a1.q[1] = *(const uint4*)&As[cur][wm + 16 + r][16 + 8 * hi];
    b0.q[0] = *(const uint4*)&Bs[cur][wn + r][8 * hi];       // B: K = 8*hi+t, 16+8*hi+t
    b0.q[1] = *(const uint4*)&Bs[cur][wn + r][16 + 8 * hi];
    b1.q[0] = *(const uint4*)&Bs[cur][wn + 16 + r][8 * hi];
    b1.q[1] = *(const uint4*)&Bs[cur][wn + 16 + r][16 + 8 * hi];

    acc00 = __builtin_amdgcn_wmma_f32_16x16x32_f16(
        false, a0.v, false, b0.v, (short)0, acc00, false, false);
    acc01 = __builtin_amdgcn_wmma_f32_16x16x32_f16(
        false, a0.v, false, b1.v, (short)0, acc01, false, false);
    acc10 = __builtin_amdgcn_wmma_f32_16x16x32_f16(
        false, a1.v, false, b0.v, (short)0, acc10, false, false);
    acc11 = __builtin_amdgcn_wmma_f32_16x16x32_f16(
        false, a1.v, false, b1.v, (short)0, acc11, false, false);

#if USE_ASYNC_LDS
    __builtin_amdgcn_s_wait_asynccnt(0);   // async tile complete before barrier
#endif
    __syncthreads();
    cur ^= 1;
  }

  // ---- store D: vgpr rr -> M = rr + 8*(lane/16), N = lane%16 ----
  const int mb0 = m0 + wm + (lane >> 4) * 8;   // acc0x rows
  const int mb1 = mb0 + 16;                    // acc1x rows
  const int na  = n0 + wn + (lane & 15);
  const int nb  = na + 16;
  if (mb0 < Cout && na < Np) {
#pragma unroll
    for (int rr = 0; rr < 8; ++rr)
      y[(size_t)(mb0 + rr) * Np + na] = acc00[rr];
  }
  if (mb0 < Cout && nb < Np) {
#pragma unroll
    for (int rr = 0; rr < 8; ++rr)
      y[(size_t)(mb0 + rr) * Np + nb] = acc01[rr];
  }
  if (mb1 < Cout && na < Np) {
#pragma unroll
    for (int rr = 0; rr < 8; ++rr)
      y[(size_t)(mb1 + rr) * Np + na] = acc10[rr];
  }
  if (mb1 < Cout && nb < Np) {
#pragma unroll
    for (int rr = 0; rr < 8; ++rr)
      y[(size_t)(mb1 + rr) * Np + nb] = acc11[rr];
  }
}

// ---------------------------------------------------------------------------
// BatchNorm statistics (population mean/var per channel) + fused apply.
// ---------------------------------------------------------------------------
__global__ __launch_bounds__(256) void bn_stats_kernel(
    const float* __restrict__ x, float* __restrict__ mean,
    float* __restrict__ var, int Np)
{
  __shared__ float s1[256], s2[256];
  int c = blockIdx.x;
  const float* p = x + (size_t)c * Np;
  float a = 0.f, b = 0.f;
  for (int i = threadIdx.x; i < Np; i += 256) {
    float v = p[i];
    a += v; b += v * v;
  }
  s1[threadIdx.x] = a; s2[threadIdx.x] = b;
  __syncthreads();
  for (int o = 128; o > 0; o >>= 1) {
    if (threadIdx.x < (unsigned)o) {
      s1[threadIdx.x] += s1[threadIdx.x + o];
      s2[threadIdx.x] += s2[threadIdx.x + o];
    }
    __syncthreads();
  }
  if (threadIdx.x == 0) {
    float m = s1[0] / (float)Np;
    mean[c] = m;
    var[c]  = s2[0] / (float)Np - m * m;
  }
}

__global__ __launch_bounds__(256) void bn_relu_kernel(
    const float* __restrict__ x, float* __restrict__ y,
    const float* __restrict__ mean, const float* __restrict__ var,
    const float* __restrict__ g, const float* __restrict__ b,
    int Np, int total)
{
  int i = blockIdx.x * blockDim.x + threadIdx.x;
  if (i >= total) return;
  int c = i / Np;
  float t = (x[i] - mean[c]) * rsqrtf(var[c] + 1e-5f) * g[c] + b[c];
  y[i] = t > 0.f ? t : 0.f;
}

// ---------------------------------------------------------------------------
// 2x bilinear upsample matching the reference's interpolation.
// ---------------------------------------------------------------------------
__global__ __launch_bounds__(256) void upsample2x_kernel(
    const float* __restrict__ x, float* __restrict__ y,
    int C, int Hin, int Win)
{
  int Ho = Hin * 2, Wo = Win * 2;
  int total = C * Ho * Wo;
  int i = blockIdx.x * blockDim.x + threadIdx.x;
  if (i >= total) return;
  int c   = i / (Ho * Wo);
  int rem = i - c * Ho * Wo;
  int oy  = rem / Wo;
  int ox  = rem - oy * Wo;
  float py = (float)oy * (float)(Hin - 1) / (float)(Ho - 1);
  float px = (float)ox * (float)(Win - 1) / (float)(Wo - 1);
  int y0 = (int)floorf(py); int y1 = y0 + 1 < Hin ? y0 + 1 : Hin - 1;
  int x0 = (int)floorf(px); int x1 = x0 + 1 < Win ? x0 + 1 : Win - 1;
  float wy = py - (float)y0;
  float wx = px - (float)x0;
  const float* p = x + (size_t)c * Hin * Win;
  float v00 = p[y0 * Win + x0], v01 = p[y0 * Win + x1];
  float v10 = p[y1 * Win + x0], v11 = p[y1 * Win + x1];
  float top = v00 * (1.f - wx) + v01 * wx;
  float bot = v10 * (1.f - wx) + v11 * wx;
  y[i] = top * (1.f - wy) + bot * wy;
}

// ---------------------------------------------------------------------------
// Disparity head: 3x3 conv (Cout=1) + bias + sigmoid, scaled by 2.
// ---------------------------------------------------------------------------
__global__ __launch_bounds__(256) void disp_kernel(
    const float* __restrict__ x, const float* __restrict__ w,
    const float* __restrict__ bias, float* __restrict__ out,
    int Cin, int Hh, int Ww)
{
  int Np = Hh * Ww;
  int i = blockIdx.x * blockDim.x + threadIdx.x;
  if (i >= Np) return;
  int oy = i / Ww, ox = i - oy * Ww;
  float acc = bias[0];
  for (int c = 0; c < Cin; ++c) {
    const float* xp = x + (size_t)c * Np;
    const float* wp = w + (size_t)c * 9;
#pragma unroll
    for (int fy = 0; fy < 3; ++fy) {
      int iy = oy - 1 + fy;
      if (iy < 0 || iy >= Hh) continue;
#pragma unroll
      for (int fx = 0; fx < 3; ++fx) {
        int ix = ox - 1 + fx;
        if (ix < 0 || ix >= Ww) continue;
        acc += xp[iy * Ww + ix] * wp[fy * 3 + fx];
      }
    }
  }
  out[i] = 2.0f / (1.0f + expf(-acc));
}

// ---------------------------------------------------------------------------
// Host-side orchestration
// ---------------------------------------------------------------------------
static void run_conv(const float* x, int Cin, int Hin, int Win,
                     const float* w, int Cout, int ks, int stride,
                     float* y, int Hout, int Wout,
                     _Float16* x16, _Float16* w16, hipStream_t s)
{
  int pad   = (ks - 1) / 2;
  int Np    = Hout * Wout;
  int K     = Cin * ks * ks;
  int Kp    = (K + 31) & ~31;
  int Coutp = (Cout + CT_M - 1) & ~(CT_M - 1);

  long nw = (long)Coutp * Kp;
  pack_weights_f16_kernel<<<(int)((nw + 255) / 256), 256, 0, s>>>(w, w16, Cout, K, Kp, nw);
  long nx = (long)Cin * Hin * Win;
  cvt_f16_kernel<<<(int)((nx + 255) / 256), 256, 0, s>>>(x, x16, nx);

  dim3 grid((Np + CT_N - 1) / CT_N, Coutp / CT_M);
  switch (ks) {
    case 3:
      conv_igemm_wmma_kernel<3><<<grid, 256, 0, s>>>(x16, w16, y, Cin, Hin, Win,
                                                     Hout, Wout, stride, pad, K, Kp, Cout);
      break;
    case 5:
      conv_igemm_wmma_kernel<5><<<grid, 256, 0, s>>>(x16, w16, y, Cin, Hin, Win,
                                                     Hout, Wout, stride, pad, K, Kp, Cout);
      break;
    default:
      conv_igemm_wmma_kernel<7><<<grid, 256, 0, s>>>(x16, w16, y, Cin, Hin, Win,
                                                     Hout, Wout, stride, pad, K, Kp, Cout);
      break;
  }
}

static void run_bn_relu(float* t, int C, int Np, const float* g, const float* b,
                        float* mean, float* var, hipStream_t s)
{
  bn_stats_kernel<<<C, 256, 0, s>>>(t, mean, var, Np);
  int total = C * Np;
  bn_relu_kernel<<<(total + 255) / 256, 256, 0, s>>>(t, t, mean, var, g, b, Np, total);
}

static void run_upsample(const float* x, float* y, int C, int Hin, int Win,
                         hipStream_t s)
{
  int total = C * 4 * Hin * Win;
  upsample2x_kernel<<<(total + 255) / 256, 256, 0, s>>>(x, y, C, Hin, Win);
}

// Param leaf indices (JAX pytree: dicts flatten in sorted-key order), after
// the 6 array inputs. down={b1,b2,g1,g2,w1,w2}; cbr={b,g,w}; disp={bias,w}.
#define I_CONV1 6
#define I_CONV2 12
#define I_CONV3 18
#define I_CONV4 24
#define I_CONV5 30
#define I_DISP0 36
#define I_DISP1 38
#define I_DISP2 40
#define I_DISP3 42
#define I_ICONV0 44
#define I_ICONV1 47
#define I_ICONV2 50
#define I_ICONV3 53
#define I_ICONV4 56
#define I_UPCONV0 59
#define I_UPCONV1 62
#define I_UPCONV2 65
#define I_UPCONV3 68
#define I_UPCONV4 71

extern "C" void kernel_launch(void* const* d_in, const int* in_sizes, int n_in,
                              void* d_out, int out_size, void* d_ws, size_t ws_size,
                              hipStream_t stream)
{
  (void)in_sizes; (void)n_in; (void)out_size; (void)ws_size;
  const float* ref  = (const float*)d_in[0];
  const float* src  = (const float*)d_in[1];
  const float* KR   = (const float*)d_in[2];
  const float* Kt   = (const float*)d_in[3];
  const float* Aamb = (const float*)d_in[4];
  const float* beta = (const float*)d_in[5];
  float* out = (float*)d_out;
  float* ws  = (float*)d_ws;

  auto P = [&](int i) -> const float* { return (const float*)d_in[i]; };

  // bump allocator over workspace (floats)
  size_t off = 0;
  auto alloc = [&](size_t n) { float* p = ws + off; off += n; return p; };
  float* xcat = alloc((size_t)259 * NP0);   // [ref(3) | cv(256)] at level 0
  float* tbuf = alloc((size_t)128 * NP0);   // pre-stride temporaries (down path)
  float* c1   = alloc((size_t)128 * NP1);
  float* c2   = alloc((size_t)256 * NP2);
  float* c3   = alloc((size_t)512 * NP3);
  float* c4   = alloc((size_t)512 * NP4);
  float* c5   = alloc((size_t)512 * NP5);
  float* ubuf = alloc((size_t)128 * NP0);   // upsample outputs
  float* vbuf = alloc((size_t)64 * NP0);    // upconv outputs
  float* catb = alloc((size_t)65 * NP0);    // concat staging
  float* ibuf = alloc((size_t)64 * NP0);    // iconv outputs
  float* udb  = alloc((size_t)NP0);         // upsampled disparity
  float* mean = alloc(1024);
  float* var  = alloc(1024);
  _Float16* w16 = (_Float16*)alloc(2400000);   // padded f16 weights (max iconv4)
  _Float16* x16 = (_Float16*)alloc(6365184);   // f16 activations (max xcat)

  const size_t F = sizeof(float);

  // -------- cost volume + concat --------
  (void)hipMemcpyAsync(xcat, ref, (size_t)3 * NP0 * F, hipMemcpyDeviceToDevice, stream);
  {
    long total = (long)NSTEPS * NP0;
    cost_volume_kernel<<<(int)((total + 255) / 256), 256, 0, stream>>>(
        ref, src, KR, Kt, Aamb, beta, xcat + (size_t)3 * NP0);
  }

  // -------- encoder --------
  run_conv(xcat, 259, HH, WW, P(I_CONV1 + 4), 128, 7, 1, tbuf, HH, WW, x16, w16, stream);
  run_bn_relu(tbuf, 128, NP0, P(I_CONV1 + 2), P(I_CONV1 + 0), mean, var, stream);
  run_conv(tbuf, 128, HH, WW, P(I_CONV1 + 5), 128, 7, 2, c1, 96, 128, x16, w16, stream);
  run_bn_relu(c1, 128, NP1, P(I_CONV1 + 3), P(I_CONV1 + 1), mean, var, stream);

  run_conv(c1, 128, 96, 128, P(I_CONV2 + 4), 256, 5, 1, tbuf, 96, 128, x16, w16, stream);
  run_bn_relu(tbuf, 256, NP1, P(I_CONV2 + 2), P(I_CONV2 + 0), mean, var, stream);
  run_conv(tbuf, 256, 96, 128, P(I_CONV2 + 5), 256, 5, 2, c2, 48, 64, x16, w16, stream);
  run_bn_relu(c2, 256, NP2, P(I_CONV2 + 3), P(I_CONV2 + 1), mean, var, stream);

  run_conv(c2, 256, 48, 64, P(I_CONV3 + 4), 512, 3, 1, tbuf, 48, 64, x16, w16, stream);
  run_bn_relu(tbuf, 512, NP2, P(I_CONV3 + 2), P(I_CONV3 + 0), mean, var, stream);
  run_conv(tbuf, 512, 48, 64, P(I_CONV3 + 5), 512, 3, 2, c3, 24, 32, x16, w16, stream);
  run_bn_relu(c3, 512, NP3, P(I_CONV3 + 3), P(I_CONV3 + 1), mean, var, stream);

  run_conv(c3, 512, 24, 32, P(I_CONV4 + 4), 512, 3, 1, tbuf, 24, 32, x16, w16, stream);
  run_bn_relu(tbuf, 512, NP3, P(I_CONV4 + 2), P(I_CONV4 + 0), mean, var, stream);
  run_conv(tbuf, 512, 24, 32, P(I_CONV4 + 5), 512, 3, 2, c4, 12, 16, x16, w16, stream);
  run_bn_relu(c4, 512, NP4, P(I_CONV4 + 3), P(I_CONV4 + 1), mean, var, stream);

  run_conv(c4, 512, 12, 16, P(I_CONV5 + 4), 512, 3, 1, tbuf, 12, 16, x16, w16, stream);
  run_bn_relu(tbuf, 512, NP4, P(I_CONV5 + 2), P(I_CONV5 + 0), mean, var, stream);
  run_conv(tbuf, 512, 12, 16, P(I_CONV5 + 5), 512, 3, 2, c5, 6, 8, x16, w16, stream);
  run_bn_relu(c5, 512, NP5, P(I_CONV5 + 3), P(I_CONV5 + 1), mean, var, stream);

  // -------- decoder --------
  run_upsample(c5, ubuf, 512, 6, 8, stream);
  run_conv(ubuf, 512, 12, 16, P(I_UPCONV4 + 2), 512, 3, 1, vbuf, 12, 16, x16, w16, stream);
  run_bn_relu(vbuf, 512, NP4, P(I_UPCONV4 + 1), P(I_UPCONV4 + 0), mean, var, stream);
  (void)hipMemcpyAsync(catb, vbuf, (size_t)512 * NP4 * F, hipMemcpyDeviceToDevice, stream);
  (void)hipMemcpyAsync(catb + (size_t)512 * NP4, c4, (size_t)512 * NP4 * F, hipMemcpyDeviceToDevice, stream);
  run_conv(catb, 1024, 12, 16, P(I_ICONV4 + 2), 512, 3, 1, ibuf, 12, 16, x16, w16, stream);
  run_bn_relu(ibuf, 512, NP4, P(I_ICONV4 + 1), P(I_ICONV4 + 0), mean, var, stream);

  run_upsample(ibuf, ubuf, 512, 12, 16, stream);
  run_conv(ubuf, 512, 24, 32, P(I_UPCONV3 + 2), 512, 3, 1, vbuf, 24, 32, x16, w16, stream);
  run_bn_relu(vbuf, 512, NP3, P(I_UPCONV3 + 1), P(I_UPCONV3 + 0), mean, var, stream);
  (void)hipMemcpyAsync(catb, vbuf, (size_t)512 * NP3 * F, hipMemcpyDeviceToDevice, stream);
  (void)hipMemcpyAsync(catb + (size_t)512 * NP3, c3, (size_t)512 * NP3 * F, hipMemcpyDeviceToDevice, stream);
  run_conv(catb, 1024, 24, 32, P(I_ICONV3 + 2), 512, 3, 1, ibuf, 24, 32, x16, w16, stream);
  run_bn_relu(ibuf, 512, NP3, P(I_ICONV3 + 1), P(I_ICONV3 + 0), mean, var, stream);

  float* d3 = out + NP0 + NP1 + NP2;
  disp_kernel<<<(NP3 + 255) / 256, 256, 0, stream>>>(ibuf, P(I_DISP3 + 1), P(I_DISP3 + 0), d3, 512, 24, 32);
  run_upsample(d3, udb, 1, 24, 32, stream);

  run_upsample(ibuf, ubuf, 512, 24, 32, stream);
  run_conv(ubuf, 512, 48, 64, P(I_UPCONV2 + 2), 256, 3, 1, vbuf, 48, 64, x16, w16, stream);
  run_bn_relu(vbuf, 256, NP2, P(I_UPCONV2 + 1), P(I_UPCONV2 + 0), mean, var, stream);
  (void)hipMemcpyAsync(catb, vbuf, (size_t)256 * NP2 * F, hipMemcpyDeviceToDevice, stream);
  (void)hipMemcpyAsync(catb + (size_t)256 * NP2, c2, (size_t)256 * NP2 * F, hipMemcpyDeviceToDevice, stream);
  (void)hipMemcpyAsync(catb + (size_t)512 * NP2, udb, (size_t)NP2 * F, hipMemcpyDeviceToDevice, stream);
  run_conv(catb, 513, 48, 64, P(I_ICONV2 + 2), 256, 3, 1, ibuf, 48, 64, x16, w16, stream);
  run_bn_relu(ibuf, 256, NP2, P(I_ICONV2 + 1), P(I_ICONV2 + 0), mean, var, stream);

  float* d2 = out + NP0 + NP1;
  disp_kernel<<<(NP2 + 255) / 256, 256, 0, stream>>>(ibuf, P(I_DISP2 + 1), P(I_DISP2 + 0), d2, 256, 48, 64);
  run_upsample(d2, udb, 1, 48, 64, stream);

  run_upsample(ibuf, ubuf, 256, 48, 64, stream);
  run_conv(ubuf, 256, 96, 128, P(I_UPCONV1 + 2), 128, 3, 1, vbuf, 96, 128, x16, w16, stream);
  run_bn_relu(vbuf, 128, NP1, P(I_UPCONV1 + 1), P(I_UPCONV1 + 0), mean, var, stream);
  (void)hipMemcpyAsync(catb, vbuf, (size_t)128 * NP1 * F, hipMemcpyDeviceToDevice, stream);
  (void)hipMemcpyAsync(catb + (size_t)128 * NP1, c1, (size_t)128 * NP1 * F, hipMemcpyDeviceToDevice, stream);
  (void)hipMemcpyAsync(catb + (size_t)256 * NP1, udb, (size_t)NP1 * F, hipMemcpyDeviceToDevice, stream);
  run_conv(catb, 257, 96, 128, P(I_ICONV1 + 2), 128, 3, 1, ibuf, 96, 128, x16, w16, stream);
  run_bn_relu(ibuf, 128, NP1, P(I_ICONV1 + 1), P(I_ICONV1 + 0), mean, var, stream);

  float* d1 = out + NP0;
  disp_kernel<<<(NP1 + 255) / 256, 256, 0, stream>>>(ibuf, P(I_DISP1 + 1), P(I_DISP1 + 0), d1, 128, 96, 128);
  run_upsample(d1, udb, 1, 96, 128, stream);

  run_upsample(ibuf, ubuf, 128, 96, 128, stream);
  run_conv(ubuf, 128, HH, WW, P(I_UPCONV0 + 2), 64, 3, 1, vbuf, HH, WW, x16, w16, stream);
  run_bn_relu(vbuf, 64, NP0, P(I_UPCONV0 + 1), P(I_UPCONV0 + 0), mean, var, stream);
  (void)hipMemcpyAsync(catb, vbuf, (size_t)64 * NP0 * F, hipMemcpyDeviceToDevice, stream);
  (void)hipMemcpyAsync(catb + (size_t)64 * NP0, udb, (size_t)NP0 * F, hipMemcpyDeviceToDevice, stream);
  run_conv(catb, 65, HH, WW, P(I_ICONV0 + 2), 64, 3, 1, ibuf, HH, WW, x16, w16, stream);
  run_bn_relu(ibuf, 64, NP0, P(I_ICONV0 + 1), P(I_ICONV0 + 0), mean, var, stream);

  disp_kernel<<<(NP0 + 255) / 256, 256, 0, stream>>>(ibuf, P(I_DISP0 + 1), P(I_DISP0 + 0), out, 64, HH, WW);
}
